// MultiHeadCrossAttention_16664473108975
// MI455X (gfx1250) — matrix operand order repaired and checked by reading, not compiled, not run on hardware
//
#include <hip/hip_runtime.h>
#include <hip/hip_bf16.h>

// ---------------- types ----------------
typedef _Float16 half16_t __attribute__((ext_vector_type(16)));
typedef _Float16 half8_t  __attribute__((ext_vector_type(8)));
typedef float    f32x8    __attribute__((ext_vector_type(8)));
typedef float    f32x4    __attribute__((ext_vector_type(4)));

union H16 { half16_t v; half8_t h8[2]; _Float16 h[16]; };

// D = A(16x32 f16) * B(32x16 f16) + C(16x16 f32)
__device__ __forceinline__ f32x8 wmma_f16(half16_t a, half16_t b, f32x8 c) {
  return __builtin_amdgcn_wmma_f32_16x16x32_f16(false, a, false, b, (short)0, c,
                                                false, false);
}

// A-fragment (16x32, 16-bit): lane m (lo half) holds K {0..7,16..23},
// lane m+16 holds K {8..15,24..31}; 8 consecutive halves per 16B load.
__device__ __forceinline__ half16_t load_a_f16(const _Float16* row, int c32, int hi) {
  H16 r;
  r.h8[0] = *(const half8_t*)(row + c32 + hi * 8);
  r.h8[1] = *(const half8_t*)(row + c32 + 16 + hi * 8);
  return r.v;
}

// B-fragment (32x16, 16-bit): lane n (lo half) holds K {0..7,16..23},
// lane n+16 holds K {8..15,24..31}. rowBT = pointer to row n of B^T (K contiguous).
__device__ __forceinline__ half16_t load_b_f16(const _Float16* rowBT, int c32, int hi) {
  H16 r;
  const _Float16* p = rowBT + c32 + hi * 8;
  r.h8[0] = *(const half8_t*)(p);
  r.h8[1] = *(const half8_t*)(p + 16);
  return r.v;
}

// A-fragment converted from an f32 row on the fly.
__device__ __forceinline__ half16_t load_a_f32cvt(const float* row, int c32, int hi) {
  H16 r;
  const float* p0 = row + c32 + hi * 8;
  const float* p1 = row + c32 + 16 + hi * 8;
  f32x4 a0 = *(const f32x4*)(p0);
  f32x4 a1 = *(const f32x4*)(p0 + 4);
  f32x4 a2 = *(const f32x4*)(p1);
  f32x4 a3 = *(const f32x4*)(p1 + 4);
#pragma unroll
  for (int i = 0; i < 4; ++i) {
    r.h[i]      = (_Float16)a0[i];
    r.h[4 + i]  = (_Float16)a1[i];
    r.h[8 + i]  = (_Float16)a2[i];
    r.h[12 + i] = (_Float16)a3[i];
  }
  return r.v;
}

// generic shared-pointer -> LDS byte offset (generic->local is low-32 truncation)
__device__ __forceinline__ unsigned lds_byte_off(const void* p) {
  return (unsigned)(uintptr_t)p;
}

// ---------------- weight transpose + f32->f16 convert ----------------
// W[k][n] (1024x1024 f32) -> WT[n][k] f16
__global__ void wt_kernel(const float* __restrict__ W, _Float16* __restrict__ WT) {
  __shared__ float tile[32][33];
  int bx = blockIdx.x, by = blockIdx.y;
  int tx = threadIdx.x, ty = threadIdx.y;  // 32 x 8
#pragma unroll
  for (int i = 0; i < 32; i += 8)
    tile[ty + i][tx] = W[(size_t)(by * 32 + ty + i) * 1024 + bx * 32 + tx];
  __syncthreads();
#pragma unroll
  for (int i = 0; i < 32; i += 8)
    WT[(size_t)(bx * 32 + ty + i) * 1024 + by * 32 + tx] = (_Float16)tile[tx][ty + i];
}

// ---------------- input projection GEMM ----------------
// C[M,1024] = X[M,1024] (f32) @ W (via WT f16) + bias, scattered to f16 dst.
// mode 0: dst[((b*16+h)*S + s)*64 + d]   (q / k layout)
// mode 1: dst[((b*16+h)*64 + d)*S + s]   (V transposed layout)
__global__ void proj_kernel(const float* __restrict__ X,
                            const _Float16* __restrict__ WT,
                            const float* __restrict__ bias,
                            _Float16* __restrict__ dst,
                            int M, int S, int mode) {
  const int K = 1024;
  int wid  = (blockIdx.x * blockDim.x + threadIdx.x) >> 5;
  int lane = threadIdx.x & 31;
  int hi = lane >> 4, ln = lane & 15;
  int mt = wid >> 4;        // M tile (16 rows)
  int nt = wid & 15;        // N strip (64 cols) == head index
  if (mt * 16 >= M) return;

  const float* arow = X + (size_t)(mt * 16 + ln) * K;
  f32x8 acc[4] = {};
  for (int c = 0; c < K; c += 32) {
    __builtin_prefetch(arow + c + 128, 0, 1);
    half16_t a = load_a_f32cvt(arow, c, hi);
#pragma unroll
    for (int t = 0; t < 4; ++t) {
      const _Float16* brow = WT + (size_t)(nt * 64 + t * 16 + ln) * K;
      acc[t] = wmma_f16(a, load_b_f16(brow, c, hi), acc[t]);
    }
  }

  int h = nt;
  int b = (mt * 16) / S;
  int sbase = (mt * 16) % S;
#pragma unroll
  for (int t = 0; t < 4; ++t) {
    int d = t * 16 + ln;                 // C layout: lane holds one column
    float bv = bias[h * 64 + d];
#pragma unroll
    for (int r = 0; r < 8; ++r) {        // rows r (lo) / r+8 (hi)
      int s = sbase + r + hi * 8;
      float val = acc[t][r] + bv;
      size_t addr = (mode == 0)
          ? (((size_t)(b * 16 + h) * S + s) * 64 + d)
          : (((size_t)(b * 16 + h) * 64 + d) * S + s);
      dst[addr] = (_Float16)val;
    }
  }
}

// ---------------- flash attention ----------------
// Block = 8 waves, all on the same (b,h); consecutive q tiles. K/V tiles are
// staged cooperatively into LDS with global_load_async_to_lds_b128 (ASYNCcnt),
// double-buffered so the async engine fetches tile i+1 while WMMAs run on i.
__global__ void attn_kernel(const _Float16* __restrict__ qf,
                            const _Float16* __restrict__ kf,
                            const _Float16* __restrict__ vT,
                            const int* __restrict__ key_mask,
                            _Float16* __restrict__ ctx) {
  const int Sq = 1024, Sk = 4096, D = 64;
  __shared__ _Float16 kbuf[2][32 * 64];  // K tile: [key][d]
  __shared__ _Float16 vbuf[2][64 * 32];  // V^T tile: [d][k]

  int tid = threadIdx.x;                 // 0..255
  int w = tid >> 5;                      // wave in block: 0..7
  int lane = tid & 31;
  int hi = lane >> 4, ln = lane & 15;
  int bh = blockIdx.x >> 3;              // b*16 + h (shared by whole block)
  int qg = blockIdx.x & 7;
  int h = bh & 15, b = bh >> 4;
  int qt = qg * 8 + w;                   // this wave's 16-query tile
  const _Float16* qh = qf + (size_t)bh * Sq * D;
  const _Float16* kh = kf + (size_t)bh * Sk * D;
  const _Float16* vh = vT + (size_t)bh * D * Sk;
  const int* mk = key_mask + b * Sk;
  int qbase = qt * 16;

  // cooperative async stage of one 32-key tile: K 4KB + V^T 4KB, 16B per thread
  auto issue_tile = [&](int kb, int bufi) {
    unsigned ldsK = lds_byte_off(&kbuf[bufi][0]) + tid * 16;
    const _Float16* gK = kh + (size_t)(kb + (tid >> 3)) * 64 + (tid & 7) * 8;
    asm volatile("global_load_async_to_lds_b128 %0, %1, off"
                 :: "v"(ldsK), "v"(gK) : "memory");
    unsigned ldsV = lds_byte_off(&vbuf[bufi][0]) + tid * 16;
    const _Float16* gV = vh + (size_t)(tid >> 2) * Sk + kb + (tid & 3) * 8;
    asm volatile("global_load_async_to_lds_b128 %0, %1, off"
                 :: "v"(ldsV), "v"(gV) : "memory");
  };

  // Q as B-fragments of S^T = K * Q^T  (two 32-wide d chunks), kept in VGPRs
  const _Float16* qrow = qh + (size_t)(qbase + ln) * D;
  half16_t q0 = load_b_f16(qrow, 0, hi);
  half16_t q1 = load_b_f16(qrow, 32, hi);

  f32x8 o[4] = {};
  float m_run = -3.0e38f, l_run = 0.0f;

  issue_tile(0, 0);
  for (int kb = 0; kb < Sk; kb += 32) {
    int cur = (kb >> 5) & 1;
    if (kb + 32 < Sk) {
      __syncthreads();                       // next buffer free for overwrite
      issue_tile(kb + 32, cur ^ 1);
      asm volatile("s_wait_asynccnt 0x2" ::: "memory");  // tile `cur` landed
    } else {
      asm volatile("s_wait_asynccnt 0x0" ::: "memory");
    }
    __syncthreads();                         // all waves see tile `cur`

    float mb = (mk[kb + lane] == 0) ? -1.0e9f : 0.0f;

    // S^T = K * Q^T from LDS fragments
    const _Float16* kr0 = &kbuf[cur][(size_t)ln * 64];
    const _Float16* kr1 = &kbuf[cur][(size_t)(16 + ln) * 64];
    f32x8 st0 = {}, st1 = {};
    st0 = wmma_f16(load_a_f16(kr0, 0, hi), q0, st0);
    st0 = wmma_f16(load_a_f16(kr0, 32, hi), q1, st0);
    st1 = wmma_f16(load_a_f16(kr1, 0, hi), q0, st1);
    st1 = wmma_f16(load_a_f16(kr1, 32, hi), q1, st1);

    // S^T layout: lane holds query column q=ln; VGPR r -> key r+8*hi (+16 st1)
    float p0[8], p1[8];
    float tmax = -3.0e38f;
#pragma unroll
    for (int r = 0; r < 8; ++r) {
      float b0 = __shfl(mb, r + hi * 8, 32);
      float b1 = __shfl(mb, 16 + r + hi * 8, 32);
      p0[r] = st0[r] * 0.125f + b0;   // scale = 1/sqrt(64)
      p1[r] = st1[r] * 0.125f + b1;
      tmax = fmaxf(tmax, fmaxf(p0[r], p1[r]));
    }
    tmax = fmaxf(tmax, __shfl_xor(tmax, 16, 32));
    float m_new = fmaxf(m_run, tmax);
    float corr = __expf(m_run - m_new);
    float tsum = 0.0f;
    H16 pa;
#pragma unroll
    for (int r = 0; r < 8; ++r) {
      p0[r] = __expf(p0[r] - m_new);
      p1[r] = __expf(p1[r] - m_new);
      tsum += p0[r] + p1[r];
      // exp(S^T) is already in the A-fragment layout of P(16q x 32k).
      pa.h[r]     = (_Float16)p0[r];
      pa.h[8 + r] = (_Float16)p1[r];
    }
    tsum += __shfl_xor(tsum, 16, 32);
    l_run = l_run * corr + tsum;
    m_run = m_new;

#pragma unroll
    for (int r = 0; r < 8; ++r) {
      float cf = __shfl(corr, r + hi * 8, 32);
#pragma unroll
      for (int t = 0; t < 4; ++t) o[t][r] *= cf;
    }

    // O += P @ V from LDS V^T fragments
#pragma unroll
    for (int t = 0; t < 4; ++t) {
      const _Float16* vrow = &vbuf[cur][(size_t)(t * 16 + ln) * 32];
      o[t] = wmma_f16(pa.v, load_b_f16(vrow, 0, hi), o[t]);
    }
  }

  // normalize and store ctx[b][s][h*64+d] as f16
#pragma unroll
  for (int r = 0; r < 8; ++r) {
    float lf = __shfl(l_run, r + hi * 8, 32);
    float inv = 1.0f / lf;
    int s = qbase + r + hi * 8;
    size_t base = ((size_t)b * Sq + s) * 1024 + h * 64;
#pragma unroll
    for (int t = 0; t < 4; ++t)
      ctx[base + t * 16 + ln] = (_Float16)(o[t][r] * inv);
  }
}

// ---------------- output projection ----------------
__global__ void out_gemm(const _Float16* __restrict__ ctx,
                         const _Float16* __restrict__ WoT,
                         const float* __restrict__ bo,
                         float* __restrict__ out) {
  const int K = 1024;
  int wid  = (blockIdx.x * blockDim.x + threadIdx.x) >> 5;
  int lane = threadIdx.x & 31;
  int hi = lane >> 4, ln = lane & 15;
  int mt = wid >> 4, nt = wid & 15;

  const _Float16* arow = ctx + (size_t)(mt * 16 + ln) * K;
  f32x8 acc[4] = {};
  for (int c = 0; c < K; c += 32) {
    __builtin_prefetch(arow + c + 128, 0, 1);
    half16_t a = load_a_f16(arow, c, hi);
#pragma unroll
    for (int t = 0; t < 4; ++t) {
      const _Float16* brow = WoT + (size_t)(nt * 64 + t * 16 + ln) * K;
      acc[t] = wmma_f16(a, load_b_f16(brow, c, hi), acc[t]);
    }
  }
#pragma unroll
  for (int t = 0; t < 4; ++t) {
    int n = nt * 64 + t * 16 + ln;
    float bv = bo[n];
#pragma unroll
    for (int r = 0; r < 8; ++r)
      out[(size_t)(mt * 16 + r + 8 * hi) * 1024 + n] = acc[t][r] + bv;
  }
}

// ---------------- launch ----------------
extern "C" void kernel_launch(void* const* d_in, const int* in_sizes, int n_in,
                              void* d_out, int out_size, void* d_ws, size_t ws_size,
                              hipStream_t stream) {
  const float* query = (const float*)d_in[0];
  const float* key   = (const float*)d_in[1];
  const float* value = (const float*)d_in[2];
  const int*   mask  = (const int*)d_in[3];
  const float* Wq = (const float*)d_in[4];  const float* bq = (const float*)d_in[5];
  const float* Wk = (const float*)d_in[6];  const float* bk = (const float*)d_in[7];
  const float* Wv = (const float*)d_in[8];  const float* bv = (const float*)d_in[9];
  const float* Wo = (const float*)d_in[10]; const float* bo = (const float*)d_in[11];
  float* out = (float*)d_out;

  // workspace layout (f16 elements), total 24M halves = 48 MB
  _Float16* WqT = (_Float16*)d_ws;
  _Float16* WkT = WqT + (1u << 20);
  _Float16* WvT = WkT + (1u << 20);
  _Float16* WoT = WvT + (1u << 20);
  _Float16* qf  = WoT + (1u << 20);   // [2,16,1024,64]
  _Float16* kf  = qf  + (2u << 20);   // [2,16,4096,64]
  _Float16* vT  = kf  + (8u << 20);   // [2,16,64,4096]
  _Float16* ctx = vT  + (8u << 20);   // [2,1024,1024]

  dim3 tb(32, 8), tg(32, 32);
  wt_kernel<<<tg, tb, 0, stream>>>(Wq, WqT);
  wt_kernel<<<tg, tb, 0, stream>>>(Wk, WkT);
  wt_kernel<<<tg, tb, 0, stream>>>(Wv, WvT);
  wt_kernel<<<tg, tb, 0, stream>>>(Wo, WoT);

  // one wave per 16x64 output tile; 256 threads = 8 waves per block
  proj_kernel<<<256,  256, 0, stream>>>(query, WqT, bq, qf, 2048, 1024, 0);
  proj_kernel<<<1024, 256, 0, stream>>>(key,   WkT, bk, kf, 8192, 4096, 0);
  proj_kernel<<<1024, 256, 0, stream>>>(value, WvT, bv, vT, 8192, 4096, 1);

  // 8 waves per block share (b,h): 32 bh * 8 q-groups = 256 blocks
  attn_kernel<<<256, 256, 0, stream>>>(qf, kf, vT, mask, ctx);

  out_gemm<<<256, 256, 0, stream>>>(ctx, WoT, bo, out);
}
